// LiteAttention_44616120270875
// MI455X (gfx1250) — compile-verified, matrix operand order checked by reading, not run on hardware
//
#include <hip/hip_runtime.h>

// ---------------------------------------------------------------------------
// LiteAttention on MI455X (gfx1250): bf16 WMMA for QKV GEMM, flash-attention
// core (QK^T and PV), and output GEMM. f32 accumulate throughout.
// A-tile staging in the GEMM uses the Tensor Data Mover (tensor_load_to_lds)
// with hardware LDS padding; completion via s_wait_tensorcnt.
// ---------------------------------------------------------------------------

#define HID   4096
#define NH    32
#define NKV   8
#define HD    128
#define QKV_N ((NH + 2 * NKV) * HD)   // 6144

typedef __bf16 bf16x16 __attribute__((ext_vector_type(16)));
typedef float  f32x8   __attribute__((ext_vector_type(8)));
typedef unsigned int u32x4 __attribute__((ext_vector_type(4)));
typedef unsigned int u32x8 __attribute__((ext_vector_type(8)));

union Frag { uint4 u4[2]; bf16x16 v; };

__device__ __forceinline__ unsigned short f32_to_bf16(float f) {
    unsigned int u = __float_as_uint(f);
    u += 0x7FFFu + ((u >> 16) & 1u);          // round-to-nearest-even
    return (unsigned short)(u >> 16);
}

__device__ __forceinline__ f32x8 wmma_bf16(bf16x16 a, bf16x16 b, f32x8 c) {
    return __builtin_amdgcn_wmma_f32_16x16x32_bf16(
        /*neg_a=*/false, a, /*neg_b=*/false, b,
        /*c_mod=*/(short)0, c, /*reuse_a=*/false, /*reuse_b=*/false);
}

// ---------------------------------------------------------------------------
// TDM: async 2D tile copy global -> LDS (bf16 elements), optional LDS pad.
// D# group0 (4 SGPRs) + group1 (8 SGPRs); 2D tensor => VADDR2/3 = NULL form.
// pad_interval_code: pad after (2<<code) DWORDs; pad_amount_code: (code+1) DWORDs.
// ---------------------------------------------------------------------------
__device__ __forceinline__ void tdm_load_2d_bf16(
    const unsigned short* gsrc, unsigned lds_off,
    unsigned tile_w_elems, unsigned tile_h_rows, unsigned row_stride_elems,
    unsigned pad_enable, unsigned pad_interval_code, unsigned pad_amount_code) {
    unsigned long long ga = (unsigned long long)(size_t)gsrc;
    u32x4 g0;
    g0.x = 1u;                                            // count=1, user mode
    g0.y = lds_off;                                       // LDS byte address
    g0.z = (unsigned)(ga & 0xFFFFFFFFu);                  // global_addr[31:0]
    g0.w = (unsigned)((ga >> 32) & 0x01FFFFFFu)           // global_addr[56:32]
         | (2u << 30);                                    // type = 2 (image)
    u32x8 g1;
    g1.s0 = (1u << 16)                                    // data_size = 2 bytes
          | (pad_enable << 20)
          | ((pad_interval_code & 7u) << 22)
          | ((pad_amount_code & 127u) << 25);
    g1.s1 = (tile_w_elems & 0xFFFFu) << 16;               // tensor_dim0[15:0]
    g1.s2 = ((tile_w_elems >> 16) & 0xFFFFu)              // tensor_dim0[31:16]
          | ((tile_h_rows & 0xFFFFu) << 16);              // tensor_dim1[15:0]
    g1.s3 = ((tile_h_rows >> 16) & 0xFFFFu)               // tensor_dim1[31:16]
          | ((tile_w_elems & 0xFFFFu) << 16);             // tile_dim0
    g1.s4 = (tile_h_rows & 0xFFFFu);                      // tile_dim1 (tile_dim2=0)
    g1.s5 = row_stride_elems;                             // tensor_dim0_stride[31:0]
    g1.s6 = 0u;                                           // stride[47:32] | dim1_stride lo
    g1.s7 = 0u;
    asm volatile("tensor_load_to_lds %0, %1" :: "s"(g0), "s"(g1) : "memory");
}

// ---------------------------------------------------------------------------
// f32 -> bf16 bulk convert
// ---------------------------------------------------------------------------
__global__ void cvt_f32_bf16_kernel(const float* __restrict__ src,
                                    unsigned short* __restrict__ dst, int n) {
    int i = blockIdx.x * blockDim.x + threadIdx.x;
    int stride = gridDim.x * blockDim.x;
    for (; i < n; i += stride) dst[i] = f32_to_bf16(src[i]);
}

// ---------------------------------------------------------------------------
// Tiled bf16 GEMM: C[M,N] = A[M,K] * B[K,N], f32 out.
// BM=128, BN=64, BK=32. 256 threads = 8 waves, each wave 32x32 (2x2 WMMA).
// A tile staged by TDM (with hardware pad to 80B rows); B tile staged with a
// manual transpose (overlaps the in-flight TDM copy).
// ---------------------------------------------------------------------------
#define BM 128
#define BN 64
#define BK 32

__global__ __launch_bounds__(256) void gemm_bf16_kernel(
    const unsigned short* __restrict__ A, const unsigned short* __restrict__ B,
    float* __restrict__ C, int M, int N, int K) {
    __shared__ __align__(16) unsigned short As[BM][BK + 8];   // 80B row stride
    __shared__ __align__(16) unsigned short Bst[BN][BK + 8];  // B transposed [n][k]

    const int tid  = threadIdx.x;
    const int wave = tid >> 5;
    const int lane = tid & 31;
    const int lrow = lane & 15;
    const int lhi  = lane >> 4;
    const int bm   = blockIdx.y * BM;
    const int bn   = blockIdx.x * BN;
    const int wm   = (wave >> 1) * 32;     // 0,32,64,96
    const int wn   = (wave & 1) * 32;      // 0,32

    const unsigned as_off = (unsigned)(size_t)(&As[0][0]);

    f32x8 acc[2][2] = {};

    for (int k0 = 0; k0 < K; k0 += BK) {
        // A tile 128x32 via Tensor Data Mover; pad 4 DWORDs after every
        // 16 DWORDs (64B row) -> 80B LDS row stride, matching As[][40].
        if (wave == 0) {
            tdm_load_2d_bf16(A + (size_t)bm * K + k0, as_off,
                             BK, BM, (unsigned)K,
                             /*pad_enable=*/1u, /*interval 16dw=*/3u, /*amount 4dw=*/3u);
        }
        // B tile: 32(k) x 64(n), read coalesced along n, store transposed.
        for (int it = tid; it < BK * (BN / 8); it += 256) {
            int kk = it >> 3;
            int nn = (it & 7) * 8;
            uint4 d = *reinterpret_cast<const uint4*>(B + (size_t)(k0 + kk) * N + bn + nn);
            const unsigned short* ds = reinterpret_cast<const unsigned short*>(&d);
#pragma unroll
            for (int e = 0; e < 8; ++e) Bst[nn + e][kk] = ds[e];
        }
        if (k0 + BK < K) {   // prefetch next K-slab (global_prefetch_b8)
            __builtin_prefetch((const void*)(A + (size_t)(bm + (tid >> 1)) * K + k0 + BK), 0, 1);
            __builtin_prefetch((const void*)(B + (size_t)(k0 + BK + (tid & 31)) * N + bn), 0, 1);
        }
        if (wave == 0) __builtin_amdgcn_s_wait_tensorcnt(0);
        __syncthreads();

        // Fragments: A row striped per lane (K: h*16 + (lane>=16)*8 + j)
        Frag a[2], b[2];
#pragma unroll
        for (int t = 0; t < 2; ++t) {
            const unsigned short* ar = &As[wm + t * 16 + lrow][0];
            a[t].u4[0] = *reinterpret_cast<const uint4*>(ar + lhi * 8);
            a[t].u4[1] = *reinterpret_cast<const uint4*>(ar + lhi * 8 + 16);
            const unsigned short* br = &Bst[wn + t * 16 + lrow][0];
            b[t].u4[0] = *reinterpret_cast<const uint4*>(br + lhi * 8);
            b[t].u4[1] = *reinterpret_cast<const uint4*>(br + lhi * 8 + 16);
        }
#pragma unroll
        for (int i = 0; i < 2; ++i)
#pragma unroll
            for (int j = 0; j < 2; ++j)
                acc[i][j] = wmma_bf16(a[i].v, b[j].v, acc[i][j]);
        __syncthreads();
    }

    // C layout: VGPR r -> M = r + 8*(lane>=16); N = lane%16
#pragma unroll
    for (int i = 0; i < 2; ++i)
#pragma unroll
        for (int j = 0; j < 2; ++j)
#pragma unroll
            for (int r = 0; r < 8; ++r) {
                int m = bm + wm + i * 16 + lhi * 8 + r;
                int n = bn + wn + j * 16 + lrow;
                C[(size_t)m * N + n] = acc[i][j][r];
            }
}

// ---------------------------------------------------------------------------
// RoPE (NeoX) + head split, f32 qkv -> head-major bf16 Q/K/V
// q_bf: [NH][T][HD], k_bf/v_bf: [NKV][T][HD]
// ---------------------------------------------------------------------------
__global__ __launch_bounds__(256) void rope_split_kernel(
    const int* __restrict__ pos, const float* __restrict__ qkv,
    unsigned short* __restrict__ qb, unsigned short* __restrict__ kb,
    unsigned short* __restrict__ vb, int Tn) {
    const int t = blockIdx.x;
    const float pf = (float)pos[t];
    const float* row = qkv + (size_t)t * QKV_N;

    for (int idx = threadIdx.x; idx < NH * (HD / 2); idx += 256) {
        int h = idx >> 6, i = idx & 63;
        float inv = __powf(10000.0f, -(float)i / 64.0f);
        float s, c; __sincosf(pf * inv, &s, &c);
        float x1 = row[h * HD + i], x2 = row[h * HD + 64 + i];
        size_t o = ((size_t)h * Tn + t) * HD;
        qb[o + i]      = f32_to_bf16(x1 * c - x2 * s);
        qb[o + 64 + i] = f32_to_bf16(x2 * c + x1 * s);
    }
    for (int idx = threadIdx.x; idx < NKV * (HD / 2); idx += 256) {
        int h = idx >> 6, i = idx & 63;
        float inv = __powf(10000.0f, -(float)i / 64.0f);
        float s, c; __sincosf(pf * inv, &s, &c);
        float x1 = row[NH * HD + h * HD + i], x2 = row[NH * HD + h * HD + 64 + i];
        size_t o = ((size_t)h * Tn + t) * HD;
        kb[o + i]      = f32_to_bf16(x1 * c - x2 * s);
        kb[o + 64 + i] = f32_to_bf16(x2 * c + x1 * s);
    }
    for (int idx = threadIdx.x; idx < NKV * HD; idx += 256) {
        int h = idx >> 7, d = idx & 127;
        vb[((size_t)h * Tn + t) * HD + d] =
            f32_to_bf16(row[(NH + NKV) * HD + h * HD + d]);
    }
}

// ---------------------------------------------------------------------------
// Flash attention: block = (64 queries, 1 head), 4 waves x 16 query rows.
// Online softmax over 32-key blocks; S and PV both on v_wmma_f32_16x16x32_bf16.
// Output written token-major bf16 [T][HID] for the final GEMM.
// ---------------------------------------------------------------------------
__global__ __launch_bounds__(128) void flash_attn_kernel(
    const unsigned short* __restrict__ qb, const unsigned short* __restrict__ kb,
    const unsigned short* __restrict__ vb, unsigned short* __restrict__ attn,
    int Tn) {
    __shared__ __align__(16) unsigned short vt[HD][32 + 8];   // V transposed [dim][key]
    __shared__ __align__(16) unsigned short ptile[4][16][32]; // per-wave P tile

    const int head = blockIdx.y;
    const int kvh  = head >> 2;              // NH/NKV = 4
    const int qblk = blockIdx.x;             // 64-query tile
    const int tid  = threadIdx.x;
    const int wave = tid >> 5;
    const int lane = tid & 31;
    const int lrow = lane & 15;
    const int lhi  = lane >> 4;
    const int qbase = qblk * 64 + wave * 16; // this wave's first query row

    // Q fragments (16 x 128 -> 4 k-steps), contiguous 16B global loads
    Frag qf[4];
    const unsigned short* qrow = qb + ((size_t)head * Tn + qbase + lrow) * HD;
#pragma unroll
    for (int kk = 0; kk < 4; ++kk) {
        qf[kk].u4[0] = *reinterpret_cast<const uint4*>(qrow + kk * 32 + lhi * 8);
        qf[kk].u4[1] = *reinterpret_cast<const uint4*>(qrow + kk * 32 + lhi * 8 + 16);
    }

    float mrow[8], lsum[8];
#pragma unroll
    for (int r = 0; r < 8; ++r) { mrow[r] = -3.0e38f; lsum[r] = 0.0f; }
    f32x8 acc[8] = {};

    const float scale = 0.08838834764831845f;  // 1/sqrt(128)
    const int nblocks = qblk * 2 + 2;          // uniform per block -> barriers safe

    for (int jb = 0; jb < nblocks; ++jb) {
        const int kbase = jb * 32;
        __syncthreads();
        // Stage V tile transposed: 32 keys x 128 dims (coalesced global reads)
        for (int it = tid; it < 32 * HD; it += 128) {
            int kkey = it >> 7, dim = it & 127;
            vt[dim][kkey] = vb[((size_t)kvh * Tn + kbase + kkey) * HD + dim];
        }
        __syncthreads();

        // S = Q K^T (two 16-key n-tiles, 4 k-steps over head_dim)
        f32x8 s[2] = {};
#pragma unroll
        for (int nt = 0; nt < 2; ++nt) {
            const unsigned short* krow =
                kb + ((size_t)kvh * Tn + kbase + nt * 16 + lrow) * HD;
#pragma unroll
            for (int kk = 0; kk < 4; ++kk) {
                Frag kf;
                kf.u4[0] = *reinterpret_cast<const uint4*>(krow + kk * 32 + lhi * 8);
                kf.u4[1] = *reinterpret_cast<const uint4*>(krow + kk * 32 + lhi * 8 + 16);
                s[nt] = wmma_bf16(qf[kk].v, kf.v, s[nt]);
            }
        }

        // Online softmax in C-register layout (row = r + 8*lhi, col = lrow)
#pragma unroll
        for (int r = 0; r < 8; ++r) {
            int qg = qbase + lhi * 8 + r;
            float v0 = s[0][r] * scale;
            float v1 = s[1][r] * scale;
            if (kbase + lrow > qg)      v0 = -1.0e30f;   // causal mask
            if (kbase + 16 + lrow > qg) v1 = -1.0e30f;
            float mx = fmaxf(v0, v1);
#pragma unroll
            for (int off = 1; off < 16; off <<= 1)
                mx = fmaxf(mx, __shfl_xor(mx, off, 16));
            float mnew = fmaxf(mrow[r], mx);
            float al = __expf(mrow[r] - mnew);
            float p0 = __expf(v0 - mnew);
            float p1 = __expf(v1 - mnew);
            float rs = p0 + p1;
#pragma unroll
            for (int off = 1; off < 16; off <<= 1)
                rs += __shfl_xor(rs, off, 16);
            lsum[r] = lsum[r] * al + rs;
            mrow[r] = mnew;
#pragma unroll
            for (int t = 0; t < 8; ++t) acc[t][r] *= al;
            int m = lhi * 8 + r;
            ptile[wave][m][lrow]      = f32_to_bf16(p0);
            ptile[wave][m][16 + lrow] = f32_to_bf16(p1);
        }
        // wave-private LDS round-trip: LDS pipe is in-order per wave
        asm volatile("s_wait_dscnt 0" ::: "memory");

        Frag pf;
        const unsigned short* prow = &ptile[wave][lrow][0];
        pf.u4[0] = *reinterpret_cast<const uint4*>(prow + lhi * 8);
        pf.u4[1] = *reinterpret_cast<const uint4*>(prow + lhi * 8 + 16);

        // acc += P (16x32) * V (32x128): 8 dim tiles
#pragma unroll
        for (int t = 0; t < 8; ++t) {
            Frag vf;
            const unsigned short* vrow = &vt[t * 16 + lrow][0];
            vf.u4[0] = *reinterpret_cast<const uint4*>(vrow + lhi * 8);
            vf.u4[1] = *reinterpret_cast<const uint4*>(vrow + lhi * 8 + 16);
            acc[t] = wmma_bf16(pf.v, vf.v, acc[t]);
        }
    }

    // Epilogue: normalize, write token-major bf16 for output GEMM
#pragma unroll
    for (int t = 0; t < 8; ++t)
#pragma unroll
        for (int r = 0; r < 8; ++r) {
            int qg = qbase + lhi * 8 + r;
            float o = acc[t][r] / lsum[r];
            attn[(size_t)qg * HID + head * HD + t * 16 + lrow] = f32_to_bf16(o);
        }
}

// ---------------------------------------------------------------------------
// Host-side launch
// ---------------------------------------------------------------------------
extern "C" void kernel_launch(void* const* d_in, const int* in_sizes, int n_in,
                              void* d_out, int out_size, void* d_ws, size_t ws_size,
                              hipStream_t stream) {
    const int*   positions = (const int*)d_in[0];
    const float* hidden    = (const float*)d_in[1];
    const float* Wqkv      = (const float*)d_in[2];
    const float* Wo        = (const float*)d_in[3];
    float*       out       = (float*)d_out;

    const int Tn = in_sizes[1] / HID;   // 2048

    // Workspace carve-up (bf16 stored as ushort)
    char* ws = (char*)d_ws;
    size_t off = 0;
    unsigned short* hs_bf   = (unsigned short*)(ws + off); off += (size_t)Tn * HID * 2;
    unsigned short* wqkv_bf = (unsigned short*)(ws + off); off += (size_t)HID * QKV_N * 2;
    unsigned short* wo_bf   = (unsigned short*)(ws + off); off += (size_t)HID * HID * 2;
    float*          qkv_f   = (float*)(ws + off);          off += (size_t)Tn * QKV_N * 4;
    unsigned short* q_bf    = (unsigned short*)(ws + off); off += (size_t)NH  * Tn * HD * 2;
    unsigned short* k_bf    = (unsigned short*)(ws + off); off += (size_t)NKV * Tn * HD * 2;
    unsigned short* v_bf    = (unsigned short*)(ws + off); off += (size_t)NKV * Tn * HD * 2;
    unsigned short* attn_bf = (unsigned short*)(ws + off); off += (size_t)Tn * HID * 2;
    (void)ws_size; (void)n_in; (void)out_size;

    // 1) bf16 conversions
    {
        int n = Tn * HID;
        cvt_f32_bf16_kernel<<<(n + 255) / 256, 256, 0, stream>>>(hidden, hs_bf, n);
        n = HID * QKV_N;
        cvt_f32_bf16_kernel<<<(n + 255) / 256, 256, 0, stream>>>(Wqkv, wqkv_bf, n);
        n = HID * HID;
        cvt_f32_bf16_kernel<<<(n + 255) / 256, 256, 0, stream>>>(Wo, wo_bf, n);
    }

    // 2) QKV GEMM: [Tn,HID] x [HID,6144] -> f32
    {
        dim3 grid(QKV_N / BN, Tn / BM);
        gemm_bf16_kernel<<<grid, 256, 0, stream>>>(hs_bf, wqkv_bf, qkv_f, Tn, QKV_N, HID);
    }

    // 3) RoPE + head split to bf16
    rope_split_kernel<<<Tn, 256, 0, stream>>>(positions, qkv_f, q_bf, k_bf, v_bf, Tn);

    // 4) Flash attention
    {
        dim3 grid(Tn / 64, NH);
        flash_attn_kernel<<<grid, 128, 0, stream>>>(q_bf, k_bf, v_bf, attn_bf, Tn);
    }

    // 5) Output GEMM: [Tn,HID] x [HID,HID] -> f32 out
    {
        dim3 grid(HID / BN, Tn / BM);
        gemm_bf16_kernel<<<grid, 256, 0, stream>>>(attn_bf, wo_bf, out, Tn, HID, HID);
    }
}
